// MonarchSSMLayer_62302795596594
// MI455X (gfx1250) — matrix operand order repaired and checked by reading, not compiled, not run on hardware
//
#include <hip/hip_runtime.h>
#include <hip/hip_bf16.h>

typedef __attribute__((ext_vector_type(16))) _Float16 v16h;
typedef __attribute__((ext_vector_type(8)))  _Float16 v8h;
typedef __attribute__((ext_vector_type(8)))  float    v8f;

#define D_MODEL 1024
#define NSTATE  4096
#define BB      64
#define BATCH   8
#define TSTEPS  2048
#define MROWS   (BATCH*TSTEPS)      // 16384
#define SCAN_WGS 4
#define BLKS_PER_WG (BB/SCAN_WGS)   // 16 blocks per workgroup
#define SCAN_THREADS 512            // 16 waves; wave = local block

// ---------------------------------------------------------------------------
// WMMA helper: D = A(16x32 f16) x B(32x16 f16) + C(f32)
// ---------------------------------------------------------------------------
__device__ __forceinline__ v8f wmma16(v16h a, v16h b, v8f c) {
  return __builtin_amdgcn_wmma_f32_16x16x32_f16(false, a, false, b, (short)0, c,
                                                false, false);
}

// Build an A-fragment from two 8-f16 runs at p and p+8 (contiguous 16 K per lane).
__device__ __forceinline__ v16h make_af(const _Float16* p) {
  v8h lo = *(const v8h*)p;
  v8h hi = *(const v8h*)(p + 8);
  v16h r;
#pragma unroll
  for (int x = 0; x < 8; ++x) { r[x] = lo[x]; r[x + 8] = hi[x]; }
  return r;
}

// A-fragment (16x32, M=i,K=j) from a row-major 64x64 f16 matrix in LDS.
__device__ __forceinline__ v16h afrag_lds(const _Float16* m, int lane, int mt, int kc) {
  int i  = (lane & 15) + (mt << 4);
  int kb = (kc << 5) + ((lane & 16) ? 16 : 0);
  return make_af(m + i * 64 + kb);
}

// B-fragment (32x16, K=j,N=a). N = lane%16 (batch, padded 8->16 with zeros),
// K run = kc*32 + (lane<16 ? 0..15 : 16..31), contiguous 16 f16 per lane.
__device__ __forceinline__ v16h bfrag_lds(const _Float16* blk /* [a][64] */,
                                          const _Float16* zpad, int lane, int kc) {
  int a  = lane & 15;
  int jb = (kc << 5) + ((lane & 16) ? 16 : 0);
  const _Float16* p = (a < 8) ? (blk + a * 64 + jb) : zpad;
  return *(const v16h*)p;
}

// ---------------------------------------------------------------------------
// Async L2->LDS copy (CDNA5): GLOBAL_LOAD_ASYNC_TO_LDS_B128, ASYNCcnt-tracked.
// No VGPR round-trip. LDS address = low 32 bits of the generic pointer
// (ISA 10.2).
// ---------------------------------------------------------------------------
__device__ __forceinline__ void async_ld_b128(void* lds_dst, const void* gsrc) {
  unsigned ldsa = (unsigned)(size_t)lds_dst;
  asm volatile("global_load_async_to_lds_b128 %0, %1, off"
               :: "v"(ldsa), "v"(gsrc) : "memory");
}
__device__ __forceinline__ void async_wait0() {
  asm volatile("s_wait_asynccnt 0x0" ::: "memory");
}

// ---------------------------------------------------------------------------
// f32 -> f16 conversion
// ---------------------------------------------------------------------------
__global__ void cvt_f16_kernel(const float* __restrict__ in,
                               _Float16* __restrict__ out, size_t n) {
  size_t i = (size_t)blockIdx.x * blockDim.x + threadIdx.x;
  size_t s = (size_t)gridDim.x * blockDim.x;
  for (; i < n; i += s) out[i] = (_Float16)in[i];
}

// ---------------------------------------------------------------------------
// Init: zero grid-barrier counter and initial state h0
// ---------------------------------------------------------------------------
__global__ void init_kernel(unsigned* __restrict__ bar, _Float16* __restrict__ HN) {
  int i = blockIdx.x * blockDim.x + threadIdx.x;
  if (i == 0) *bar = 0u;
  int s = gridDim.x * blockDim.x;
  for (int j = i; j < BATCH * NSTATE; j += s) HN[j] = (_Float16)0.f;
}

// ---------------------------------------------------------------------------
// Cayley: X = (I+S)^-1 (I-S), S = skew - skew^T. Gauss-Jordan in LDS,
// no pivoting (singular values of I+S are >= 1). 128 workgroups (64 L + 64 R).
// ---------------------------------------------------------------------------
__global__ __launch_bounds__(256) void cayley_kernel(const float* __restrict__ Lsk,
                                                     const float* __restrict__ Rsk,
                                                     _Float16* __restrict__ Lout,
                                                     _Float16* __restrict__ Rout) {
  __shared__ float aug[64][130];   // padded row stride to dodge bank conflicts
  __shared__ float fac[64];
  __shared__ float pivinv;
  const int tid = threadIdx.x;
  const bool isR = blockIdx.x >= 64;
  const int mk = blockIdx.x & 63;
  const float* S = (isR ? Rsk : Lsk) + (size_t)mk * 4096;
  _Float16*    O = (isR ? Rout : Lout) + (size_t)mk * 4096;

  for (int idx = tid; idx < 4096; idx += 256) {
    int i = idx >> 6, j = idx & 63;
    float sv = S[i * 64 + j] - S[j * 64 + i];
    float id = (i == j) ? 1.f : 0.f;
    aug[i][j]      = id + sv;   // A = I + S
    aug[i][64 + j] = id - sv;   // B = I - S (64 RHS)
  }
  __syncthreads();
  for (int p = 0; p < 64; ++p) {
    if (tid == 0) pivinv = 1.f / aug[p][p];
    __syncthreads();
    if (tid < 128) aug[p][tid] *= pivinv;
    __syncthreads();
    if (tid < 64) fac[tid] = aug[tid][p];
    __syncthreads();
    for (int idx = tid; idx < 64 * 128; idx += 256) {
      int r = idx >> 7, c = idx & 127;
      if (r != p) aug[r][c] -= fac[r] * aug[p][c];
    }
    __syncthreads();
  }
  for (int idx = tid; idx < 4096; idx += 256) {
    int i = idx >> 6, j = idx & 63;
    O[i * 64 + j] = (_Float16)aug[i][64 + j];
  }
}

// ---------------------------------------------------------------------------
// Generic f16 WMMA GEMM:  out[M][Nn] = A[M][K] @ Bw[Nn][K]^T (+bias)(sigmoid)
// (+ D*u for the readout). 64x128 output tile per WG, 8 waves, each wave owns
// a 2x2 fragment block (2 A-frags + 2 B-frags -> 4 WMMAs per k-chunk).
// ---------------------------------------------------------------------------
template <int SIG, int OUTF16, int ADDDU>
__global__ __launch_bounds__(256) void wmma_gemm(
    const _Float16* __restrict__ A, const _Float16* __restrict__ Bw,
    const float* __restrict__ bias, float* __restrict__ Of,
    _Float16* __restrict__ Oh, int M, int Nn, int K,
    const float* __restrict__ Dv, const float* __restrict__ Uo) {
  const int lane = threadIdx.x & 31;
  const int wv   = threadIdx.x >> 5;
  const int mt0  = (wv >> 2) << 1;     // 0 or 2   (M tiles mt0, mt0+1)
  const int nt0  = (wv & 3) << 1;      // 0,2,4,6  (N tiles nt0, nt0+1)
  const size_t m0 = (size_t)blockIdx.x * 64;
  const size_t n0 = (size_t)blockIdx.y * 128;

  const size_t arow0 = m0 + (size_t)(mt0 << 4) + (lane & 15);
  const size_t arow1 = arow0 + 16;
  const size_t nrow0 = n0 + (size_t)(nt0 << 4) + (lane & 15);
  const size_t nrow1 = nrow0 + 16;
  const int aoff = (lane & 16) ? 16 : 0;
  const int boff = (lane & 16) ? 16 : 0;

  v8f c00 = {}; v8f c01 = {}; v8f c10 = {}; v8f c11 = {};
  const int kcN = K >> 5;
  for (int kc = 0; kc < kcN; ++kc) {
    size_t kb = (size_t)kc << 5;
    const _Float16* pa0 = A + arow0 * K + kb + aoff;
    const _Float16* pa1 = A + arow1 * K + kb + aoff;
    if (kc + 1 < kcN) __builtin_prefetch(pa0 + 32, 0, 1);  // global_prefetch_b8
    v16h a0 = make_af(pa0);
    v16h a1 = make_af(pa1);
    v16h b0 = *(const v16h*)(Bw + nrow0 * K + kb + boff);
    v16h b1 = *(const v16h*)(Bw + nrow1 * K + kb + boff);
    c00 = wmma16(a0, b0, c00);
    c01 = wmma16(a0, b1, c01);
    c10 = wmma16(a1, b0, c10);
    c11 = wmma16(a1, b1, c11);
  }

  v8f accs[2][2] = {{c00, c01}, {c10, c11}};
  const int nA = lane & 15;
  const int madd = (lane & 16) ? 8 : 0;
#pragma unroll
  for (int im = 0; im < 2; ++im) {
#pragma unroll
    for (int jn = 0; jn < 2; ++jn) {
      size_t n = n0 + (size_t)((nt0 + jn) << 4) + nA;
      size_t mbase = m0 + (size_t)((mt0 + im) << 4) + madd;
      float bz  = bias ? bias[n] : 0.f;
      float dvn = ADDDU ? Dv[n] : 0.f;
#pragma unroll
      for (int r = 0; r < 8; ++r) {
        size_t m = mbase + r;
        float v = accs[im][jn][r] + bz;
        if (SIG) v = 1.f / (1.f + __expf(-v));
        if (ADDDU) v += dvn * Uo[m * (size_t)Nn + n];
        if (OUTF16) Oh[m * (size_t)Nn + n] = (_Float16)v;
        else        Of[m * (size_t)Nn + n] = v;
      }
    }
  }
}

// ---------------------------------------------------------------------------
// Grid barrier among SCAN_WGS workgroups (monotone target; counter zeroed by
// init_kernel every launch -> graph-replay safe).
// ---------------------------------------------------------------------------
__device__ __forceinline__ void grid_barrier(unsigned* bar, unsigned& phase) {
  __syncthreads();
  ++phase;
  if (threadIdx.x == 0) {
    unsigned target = phase * SCAN_WGS;
    __hip_atomic_fetch_add(bar, 1u, __ATOMIC_RELEASE, __HIP_MEMORY_SCOPE_AGENT);
    while (__hip_atomic_load(bar, __ATOMIC_ACQUIRE, __HIP_MEMORY_SCOPE_AGENT) < target)
      __builtin_amdgcn_s_sleep(1);
  }
  __syncthreads();
  __threadfence();   // acquire: discard stale lines before reading peers' data
}

// ---------------------------------------------------------------------------
// Sequential scan. 4 persistent WGs (512 thr = 16 waves); WG g owns blocks
// K_g = [16g, 16g+16), whose L[k],R[k] (f16, 256 KB) stay LDS-resident.
// All 8 batches ride in one WMMA N-tile (padded to 16).
// State layout HN[a][k][e] (f16); exchange HP[a][p][q] = hr[a][q][p].
// Per-step staging uses GLOBAL_LOAD_ASYNC_TO_LDS_B128 (ASYNCcnt), so the
// critical path after each grid barrier is a pure L2->LDS DMA.
// Total static LDS ~312.5 KB (< 320 KB WGP limit) -> one WG per WGP.
// ---------------------------------------------------------------------------
__global__ __launch_bounds__(SCAN_THREADS) void scan_kernel(
    const _Float16* __restrict__ Lm, const _Float16* __restrict__ Rm,
    const float* __restrict__ G, const _Float16* __restrict__ UB,
    _Float16* __restrict__ Hbig, _Float16* __restrict__ HP,
    _Float16* __restrict__ HN, unsigned* __restrict__ bar) {
  __shared__ __align__(32) _Float16 matR[BLKS_PER_WG][64 * 64];  // 128 KB
  __shared__ __align__(32) _Float16 matL[BLKS_PER_WG][64 * 64];  // 128 KB
  __shared__ __align__(32) _Float16 Sst[BLKS_PER_WG][8][64];     //  16 KB staged state
  __shared__ __align__(32) _Float16 Tout[8][64][BLKS_PER_WG];    //  16 KB transposed out
  __shared__ __align__(32) _Float16 Ust[8][64][BLKS_PER_WG];     //  16 KB staged UB
  __shared__ __align__(32) float    Gst[8][32];                  //   1 KB gates slice
  __shared__ __align__(32) _Float16 Zpad[32];                    // zeros for N-pad

  const int tid  = threadIdx.x;
  const int lane = tid & 31;
  const int wv   = tid >> 5;               // wave = local block index (0..15)
  const int kgb  = blockIdx.x * BLKS_PER_WG;

  // Preload this WG's rotation matrices into LDS (once). 8192 v8h chunks/tensor.
  for (int c = tid; c < 8192; c += SCAN_THREADS) {
    int kl = c >> 9, off = (c & 511) << 3;
    *(v8h*)(&matR[kl][off]) = *(const v8h*)(Rm + (size_t)(kgb + kl) * 4096 + off);
    *(v8h*)(&matL[kl][off]) = *(const v8h*)(Lm + (size_t)(kgb + kl) * 4096 + off);
  }
  if (tid < 32) Zpad[tid] = (_Float16)0.f;

  unsigned phase = 0;
  grid_barrier(bar, phase);   // also covers h0 visibility from init_kernel

  for (int t = 0; t < TSTEPS; ++t) {
    // ---- async-stage state rows K_g, gate slice, UB slice (L2 -> LDS) ----
    for (int c = tid; c < 1024; c += SCAN_THREADS) {  // state: [a][kl] rows of 64
      int a = c >> 7, kl = (c >> 3) & 15, off = (c & 7) << 3;
      async_ld_b128(&Sst[kl][a][off],
                    HN + ((size_t)(a * 64 + kgb + kl) << 6) + off);
    }
    if (tid < 64) {                                   // gates: alpha/beta cols K_g
      int a = tid >> 3, h = (tid >> 2) & 1, q = tid & 3;
      async_ld_b128(&Gst[a][h * 16 + q * 4],
                    G + ((size_t)a * TSTEPS + t) * 128 + (h ? 64 : 0) + kgb + q * 4);
    }
    {                                                 // ub[a][i*64 + kgb..+16]
      int a = tid >> 6, i = tid & 63;
      const _Float16* gs = UB + ((size_t)a * TSTEPS + t) * NSTATE + i * 64 + kgb;
      async_ld_b128(&Ust[a][i][0], gs);
      async_ld_b128(&Ust[a][i][8], gs + 8);
    }
    async_wait0();
    __syncthreads();

    // ---- R stage: hr[a,k,i] = beta[a,k] * sum_j R[k,i,j] h[a,k,j] ----
    {
      const _Float16* mr = matR[wv];
      const _Float16* sb = &Sst[wv][0][0];
      const int aa = lane & 15;
      const float sc = (aa < 8) ? Gst[aa][16 + wv] : 0.f;   // beta
#pragma unroll
      for (int mt = 0; mt < 4; ++mt) {
        v8f acc = {};
#pragma unroll
        for (int kc = 0; kc < 2; ++kc)
          acc = wmma16(afrag_lds(mr, lane, mt, kc), bfrag_lds(sb, Zpad, lane, kc), acc);
        int ib = (mt << 4) + ((lane & 16) ? 8 : 0);
        if (aa < 8) {
#pragma unroll
          for (int r = 0; r < 8; ++r)
            Tout[aa][ib + r][wv] = (_Float16)(acc[r] * sc);   // transpose via LDS
        }
      }
    }
    __syncthreads();
    {                                                 // HP[a][i][kgb..+16]
      int a = tid >> 6, i = tid & 63;
      *(v16h*)(HP + ((size_t)(a * 64 + i) << 6) + kgb) = *(const v16h*)(&Tout[a][i][0]);
    }
    grid_barrier(bar, phase);

    // ---- async-stage hp rows K_g (produced by all WGs) ----
    for (int c = tid; c < 1024; c += SCAN_THREADS) {
      int a = c >> 7, kl = (c >> 3) & 15, off = (c & 7) << 3;
      async_ld_b128(&Sst[kl][a][off],
                    HP + ((size_t)(a * 64 + kgb + kl) << 6) + off);
    }
    async_wait0();
    __syncthreads();

    // ---- L stage + ub add: h_new[a][i][k] = alpha*L[k]@hp[a,k,:] + ub ----
    {
      const _Float16* ml = matL[wv];
      const _Float16* sb = &Sst[wv][0][0];
      const int aa = lane & 15;
      const float sc = (aa < 8) ? Gst[aa][wv] : 0.f;        // alpha
#pragma unroll
      for (int mt = 0; mt < 4; ++mt) {
        v8f acc = {};
#pragma unroll
        for (int kc = 0; kc < 2; ++kc)
          acc = wmma16(afrag_lds(ml, lane, mt, kc), bfrag_lds(sb, Zpad, lane, kc), acc);
        int ib = (mt << 4) + ((lane & 16) ? 8 : 0);
        if (aa < 8) {
#pragma unroll
          for (int r = 0; r < 8; ++r)
            Tout[aa][ib + r][wv] =
                (_Float16)(acc[r] * sc + (float)Ust[aa][ib + r][wv]);
        }
      }
    }
    __syncthreads();
    {                                                 // new state + Hbig row
      int a = tid >> 6, i = tid & 63;
      v16h val = *(const v16h*)(&Tout[a][i][0]);
      *(v16h*)(HN + ((size_t)(a * 64 + i) << 6) + kgb) = val;
      *(v16h*)(Hbig + ((size_t)a * TSTEPS + t) * NSTATE + i * 64 + kgb) = val;
    }
    grid_barrier(bar, phase);
  }
}

// ---------------------------------------------------------------------------
// Host launcher
// ---------------------------------------------------------------------------
extern "C" void kernel_launch(void* const* d_in, const int* in_sizes, int n_in,
                              void* d_out, int out_size, void* d_ws, size_t ws_size,
                              hipStream_t stream) {
  const float* u   = (const float*)d_in[0];
  const float* Lsk = (const float*)d_in[1];
  const float* Rsk = (const float*)d_in[2];
  const float* Wgw = (const float*)d_in[3];
  const float* Wgb = (const float*)d_in[4];
  const float* WBw = (const float*)d_in[5];
  const float* WBb = (const float*)d_in[6];
  const float* Cw  = (const float*)d_in[7];
  const float* Dv  = (const float*)d_in[8];
  float* Y = (float*)d_out;
  (void)in_sizes; (void)n_in; (void)out_size; (void)ws_size;

  char* ws = (char*)d_ws;
  size_t off = 0;
  auto take = [&](size_t bytes) {
    void* p = ws + off;
    off = (off + bytes + 255) & ~(size_t)255;
    return p;
  };
  _Float16* Uf16 = (_Float16*)take((size_t)MROWS * D_MODEL * 2);  //  32 MB
  _Float16* Wg16 = (_Float16*)take((size_t)128 * D_MODEL * 2);
  _Float16* WB16 = (_Float16*)take((size_t)NSTATE * D_MODEL * 2); //   8 MB
  _Float16* C16  = (_Float16*)take((size_t)D_MODEL * NSTATE * 2); //   8 MB
  _Float16* Lf16 = (_Float16*)take((size_t)BB * BB * BB * 2);
  _Float16* Rf16 = (_Float16*)take((size_t)BB * BB * BB * 2);
  float*    G    = (float*)take((size_t)MROWS * 128 * 4);         //   8 MB
  _Float16* UB   = (_Float16*)take((size_t)MROWS * NSTATE * 2);   // 128 MB
  _Float16* Hbig = (_Float16*)take((size_t)MROWS * NSTATE * 2);   // 128 MB
  _Float16* HP   = (_Float16*)take((size_t)BATCH * NSTATE * 2);
  _Float16* HN   = (_Float16*)take((size_t)BATCH * NSTATE * 2);
  unsigned* bar  = (unsigned*)take(256);

  init_kernel<<<dim3(16), dim3(256), 0, stream>>>(bar, HN);

  cvt_f16_kernel<<<dim3(2048), dim3(256), 0, stream>>>(u, Uf16, (size_t)MROWS * D_MODEL);
  cvt_f16_kernel<<<dim3(64),   dim3(256), 0, stream>>>(Wgw, Wg16, (size_t)128 * D_MODEL);
  cvt_f16_kernel<<<dim3(512),  dim3(256), 0, stream>>>(WBw, WB16, (size_t)NSTATE * D_MODEL);
  cvt_f16_kernel<<<dim3(512),  dim3(256), 0, stream>>>(Cw,  C16,  (size_t)D_MODEL * NSTATE);

  cayley_kernel<<<dim3(128), dim3(256), 0, stream>>>(Lsk, Rsk, Lf16, Rf16);

  // gates: sigmoid(U @ Wg^T + b), f32 out
  wmma_gemm<1, 0, 0><<<dim3(MROWS / 64, 128 / 128), dim3(256), 0, stream>>>(
      Uf16, Wg16, Wgb, G, nullptr, MROWS, 128, D_MODEL, nullptr, nullptr);
  // input injection: UB = U @ WB^T + b, f16 out
  wmma_gemm<0, 1, 0><<<dim3(MROWS / 64, NSTATE / 128), dim3(256), 0, stream>>>(
      Uf16, WB16, WBb, nullptr, UB, MROWS, NSTATE, D_MODEL, nullptr, nullptr);

  // sequential Monarch scan (persistent, 4 WGs, LDS-resident rotations)
  scan_kernel<<<dim3(SCAN_WGS), dim3(SCAN_THREADS), 0, stream>>>(
      Lf16, Rf16, G, UB, Hbig, HP, HN, bar);

  // readout: Y = H @ C^T + D * u, f32 out
  wmma_gemm<0, 0, 1><<<dim3(MROWS / 64, D_MODEL / 128), dim3(256), 0, stream>>>(
      Hbig, C16, nullptr, Y, nullptr, MROWS, D_MODEL, NSTATE, Dv, u);
}
